// InvariantPointAttention_62466004353652
// MI455X (gfx1250) — matrix operand order repaired and checked by reading, not compiled, not run on hardware
//
#include <hip/hip_runtime.h>
#include <math.h>

// Problem constants (from the reference)
#define NRES 768
#define CSD  384
#define CZD  128
#define CHD  16
#define NH   12
#define PQKD 4
#define PVD  8
#define CATD 2112                       // H*(CZ + CH + PV*4)
#define ATT_S ((long)NRES * NRES)       // per-head attention plane (elements)

typedef __attribute__((ext_vector_type(16))) _Float16 v16h;
typedef __attribute__((ext_vector_type(8)))  _Float16 v8h;
typedef __attribute__((ext_vector_type(8)))  float    v8f;
typedef __attribute__((ext_vector_type(4)))  float    v4f;

static __device__ __forceinline__ v16h cat16(v8h lo, v8h hi) {
  return __builtin_shufflevector(lo, hi, 0, 1, 2, 3, 4, 5, 6, 7,
                                 8, 9, 10, 11, 12, 13, 14, 15);
}

// ---------------------------------------------------------------------------
// CDNA5 16x16x32 f16 WMMA fragment addressing (CDNA5 ISA §7.12.2):
//  A (MxK=16x32) f16 row-major: lane -> row (lane&15); half-wave kb=8*(lane>>4)
//     needs halves [k0+kb .. +7] and [k0+16+kb .. +7]  -> two b128 loads
//  B as Bt (NxK) f16 row-major: lane -> col (lane&15); kb=8*(lane>>4)
//     needs halves [k0+kb .. +7] and [k0+16+kb .. +7]  -> two b128 loads
//  C/D f32: lane -> col (lane&15); VGPR i -> row i + 8*(lane>>4)
// ---------------------------------------------------------------------------

static __device__ __forceinline__
void store_tile(float* __restrict__ C, int ldc, int row, int col,
                int mValid, int nValid, const v8f acc, int lane,
                const float* __restrict__ bias) {
  int c = col + (lane & 15);
  if (c >= nValid) return;
  float bv = bias ? bias[c] : 0.f;
#pragma unroll
  for (int i = 0; i < 8; ++i) {
    int r = row + i + ((lane >> 4) << 3);
    if (r < mValid) C[(long)r * ldc + c] = acc[i] + bv;
  }
}

// ---------------------------------------------------------------------------
// Generic batched GEMM: C = A (MxKp f16, row-major) * Bt^T (Bt is NxKp f16)
// Kp multiple of 32 (zero-padded at pack time). One 16x16 tile per wave.
// Double-buffered K-loop: WMMA(k) overlaps the loads of k+1.
// ---------------------------------------------------------------------------
__global__ void gemm_f16_wmma(const _Float16* __restrict__ A, const _Float16* __restrict__ Bt,
                              const float* __restrict__ bias, float* __restrict__ C,
                              int M, int N, int Kp, int lda, int ldbt, int ldc,
                              long sA, long sB, long sC) {
  const _Float16* Ab = A + (long)blockIdx.y * sA;
  const _Float16* Bb = Bt + (long)blockIdx.y * sB;
  float*          Cb = C + (long)blockIdx.y * sC;
  int lane = threadIdx.x & 31, wave = threadIdx.x >> 5;
  int tn = (N + 15) >> 4, tm = (M + 15) >> 4;
  int tile = blockIdx.x * (blockDim.x >> 5) + wave;
  if (tile >= tn * tm) return;                    // wave-uniform
  int row = (tile / tn) << 4, col = (tile % tn) << 4;

  const _Float16* Ar = Ab + (long)(row + (lane & 15)) * lda + ((lane >> 4) << 3);
  const _Float16* Br = Bb + (long)(col + (lane & 15)) * ldbt + ((lane >> 4) << 3);

  v8h alo = *(const v8h*)(Ar);
  v8h ahi = *(const v8h*)(Ar + 16);
  v8h blo = *(const v8h*)(Br);
  v8h bhi = *(const v8h*)(Br + 16);
  v8f acc = {};
  for (int k0 = 32; k0 < Kp; k0 += 32) {
    v8h alo2 = *(const v8h*)(Ar + k0);
    v8h ahi2 = *(const v8h*)(Ar + k0 + 16);
    v8h blo2 = *(const v8h*)(Br + k0);
    v8h bhi2 = *(const v8h*)(Br + k0 + 16);
    __builtin_prefetch(Ar + k0 + 32, 0, 1);       // speculative: past-end is dropped
    __builtin_prefetch(Br + k0 + 32, 0, 1);
    acc = __builtin_amdgcn_wmma_f32_16x16x32_f16(false, cat16(alo, ahi),
                                                 false, cat16(blo, bhi),
                                                 (short)0, acc, false, false);
    alo = alo2; ahi = ahi2; blo = blo2; bhi = bhi2;
  }
  acc = __builtin_amdgcn_wmma_f32_16x16x32_f16(false, cat16(alo, ahi),
                                               false, cat16(blo, bhi),
                                               (short)0, acc, false, false);
  store_tile(Cb, ldc, row, col, M, N, acc, lane, bias);
}

// ---------------------------------------------------------------------------
// b_pair: att[(h,q,k)] = sqrt(1/3) * (z @ W_b + b_b)
// A = z f32 (non-temporal b128 loads + cvt), B = pre-packed WbT f16 (16x128)
// ---------------------------------------------------------------------------
__global__ void bias_pair_kernel(const float* __restrict__ z, const _Float16* __restrict__ WbT,
                                 const float* __restrict__ bb, float* __restrict__ att) {
  int lane = threadIdx.x & 31, wave = threadIdx.x >> 5;
  int tile = blockIdx.x * (blockDim.x >> 5) + wave;
  if (tile >= (NRES * NRES) / 16) return;
  int row = tile << 4;
  const float*    Zr = z + (long)(row + (lane & 15)) * CZD + ((lane >> 4) << 3);
  const _Float16* Br = WbT + (long)(lane & 15) * CZD + ((lane >> 4) << 3);
  v8f acc = {};
#pragma unroll
  for (int k0 = 0; k0 < CZD; k0 += 32) {
    v4f a0 = __builtin_nontemporal_load((const v4f*)(Zr + k0));
    v4f a1 = __builtin_nontemporal_load((const v4f*)(Zr + k0 + 4));
    v4f a2 = __builtin_nontemporal_load((const v4f*)(Zr + k0 + 16));
    v4f a3 = __builtin_nontemporal_load((const v4f*)(Zr + k0 + 20));
    v16h a;
#pragma unroll
    for (int j = 0; j < 4; ++j) {
      a[j]      = (_Float16)a0[j];
      a[4 + j]  = (_Float16)a1[j];
      a[8 + j]  = (_Float16)a2[j];
      a[12 + j] = (_Float16)a3[j];
    }
    v8h blo = *(const v8h*)(Br + k0);
    v8h bhi = *(const v8h*)(Br + k0 + 16);
    acc = __builtin_amdgcn_wmma_f32_16x16x32_f16(false, a, false, cat16(blo, bhi),
                                                 (short)0, acc, false, false);
  }
  int h = lane & 15;
  if (h < NH) {
    const float s13 = 0.57735026919f;             // sqrt(1/3)
    float bv = bb[h];
#pragma unroll
    for (int i = 0; i < 8; ++i) {
      int m = row + i + ((lane >> 4) << 3);
      int q = m / NRES, k = m % NRES;
      __builtin_nontemporal_store(s13 * (acc[i] + bv),
                                  att + ((long)h * NRES + q) * NRES + k);
    }
  }
}

// ---------------------------------------------------------------------------
// Logits: att += scale*q.k^T - 0.5*hw[h]*sum_p |q_pt-k_pt|^2 + mask
// qh/kh are f16 (h, n, 32) with K=16 zero-padded to 32 -> branch-free loads.
// ---------------------------------------------------------------------------
__global__ void logits_kernel(const _Float16* __restrict__ qh, const _Float16* __restrict__ kh,
                              const float* __restrict__ qpts, const float* __restrict__ kvpts,
                              const float* __restrict__ mask, const float* __restrict__ head_w,
                              float* __restrict__ att) {
  int lane = threadIdx.x & 31, wave = threadIdx.x >> 5;
  int h = blockIdx.y;
  const int KT = NRES / 16;                       // 48
  int t = blockIdx.x * (blockDim.x >> 5) + wave;
  if (t >= KT * KT) return;
  int qrow = (t / KT) << 4, krow = (t % KT) << 4;

  const _Float16* Aq = qh + ((long)h * NRES + qrow + (lane & 15)) * 32 + ((lane >> 4) << 3);
  const _Float16* Bk = kh + ((long)h * NRES + krow + (lane & 15)) * 32 + ((lane >> 4) << 3);
  v8h alo = *(const v8h*)(Aq);
  v8h ahi = *(const v8h*)(Aq + 16);
  v8h blo = *(const v8h*)(Bk);
  v8h bhi = *(const v8h*)(Bk + 16);
  v8f acc = {};
  acc = __builtin_amdgcn_wmma_f32_16x16x32_f16(false, cat16(alo, ahi),
                                               false, cat16(blo, bhi),
                                               (short)0, acc, false, false);

  const float scale_qk = 0.14433756729740643f;    // sqrt(1/(3*CH))
  float hwv = logf(1.f + expf(head_w[h])) * 0.13608276348795434f; // softplus*sqrt(1/54)

  int c = lane & 15;
  int kk = krow + c;
  float mk = mask[kk];
  float kp[PQKD][3];
#pragma unroll
  for (int p = 0; p < PQKD; ++p)
#pragma unroll
    for (int d = 0; d < 3; ++d)
      kp[p][d] = kvpts[(((long)kk * NH + h) * (PQKD + PVD) + p) * 3 + d];

#pragma unroll
  for (int i = 0; i < 8; ++i) {
    int qq = qrow + i + ((lane >> 4) << 3);
    float d2 = 0.f;
#pragma unroll
    for (int p = 0; p < PQKD; ++p) {
      const float* qp = qpts + (((long)qq * NH + h) * PQKD + p) * 3;
      float dx = qp[0] - kp[p][0], dy = qp[1] - kp[p][1], dz = qp[2] - kp[p][2];
      d2 += dx * dx + dy * dy + dz * dz;
    }
    long idx = ((long)h * NRES + qq) * NRES + kk;
    att[idx] += acc[i] * scale_qk - 0.5f * hwv * d2 + 100000.f * (mask[qq] * mk - 1.f);
  }
}

// ---------------------------------------------------------------------------
// o_pair[q] = a_q (12x768, f16 planes) . z_q (768x128, f32 non-temporal)
// grid.x = q ; 8 waves = 8 column tiles of 16. Double-buffered K-loop.
// ---------------------------------------------------------------------------
__global__ void opair_kernel(const _Float16* __restrict__ att16, const float* __restrict__ z,
                             float* __restrict__ opair) {
  int lane = threadIdx.x & 31, wave = threadIdx.x >> 5;
  int q = blockIdx.x;
  int col = wave << 4;
  const _Float16* Ar = att16 + (long)q * NRES + (long)(lane & 15) * ATT_S + ((lane >> 4) << 4);
  const float*    Zc = z + (long)q * NRES * CZD + (col + (lane & 15));
  int kb = (lane >> 4) << 4;

  v8h alo = *(const v8h*)(Ar);
  v8h ahi = *(const v8h*)(Ar + 8);
  v16h b;
#pragma unroll
  for (int j = 0; j < 16; ++j)
    b[j] = (_Float16)__builtin_nontemporal_load(Zc + (long)(kb + j) * CZD);

  v8f acc = {};
  for (int k0 = 32; k0 < NRES; k0 += 32) {
    v8h alo2 = *(const v8h*)(Ar + k0);
    v8h ahi2 = *(const v8h*)(Ar + k0 + 8);
    v16h b2;
#pragma unroll
    for (int j = 0; j < 16; ++j)
      b2[j] = (_Float16)__builtin_nontemporal_load(Zc + (long)(k0 + kb + j) * CZD);
    acc = __builtin_amdgcn_wmma_f32_16x16x32_f16(false, cat16(alo, ahi), false, b,
                                                 (short)0, acc, false, false);
    alo = alo2; ahi = ahi2; b = b2;
  }
  acc = __builtin_amdgcn_wmma_f32_16x16x32_f16(false, cat16(alo, ahi), false, b,
                                               (short)0, acc, false, false);

  int c = col + (lane & 15);
#pragma unroll
  for (int i = 0; i < 8; ++i) {
    int r = i + ((lane >> 4) << 3);
    if (r < NH) opair[(long)q * (NH * CZD) + r * CZD + c] = acc[i];
  }
}

// ---------------------------------------------------------------------------
// Softmax rows (length 768) of f32 att -> f16 att16 (first 12 of 16 planes)
// ---------------------------------------------------------------------------
__global__ void softmax_kernel(const float* __restrict__ att, _Float16* __restrict__ att16) {
  long row = blockIdx.x;
  const float* a = att + row * NRES;
  _Float16* o = att16 + row * NRES;
  __shared__ float red[8];
  int lane = threadIdx.x & 31, wid = threadIdx.x >> 5;

  float v0 = a[threadIdx.x], v1 = a[threadIdx.x + 256], v2 = a[threadIdx.x + 512];
  float m = fmaxf(v0, fmaxf(v1, v2));
#pragma unroll
  for (int s = 16; s; s >>= 1) m = fmaxf(m, __shfl_xor(m, s, 32));
  if (lane == 0) red[wid] = m;
  __syncthreads();
  if (threadIdx.x == 0) {
    float t = red[0];
    for (int i = 1; i < 8; ++i) t = fmaxf(t, red[i]);
    red[0] = t;
  }
  __syncthreads();
  m = red[0];
  __syncthreads();

  float e0 = __expf(v0 - m), e1 = __expf(v1 - m), e2 = __expf(v2 - m);
  float s = e0 + e1 + e2;
#pragma unroll
  for (int o2 = 16; o2; o2 >>= 1) s += __shfl_xor(s, o2, 32);
  if (lane == 0) red[wid] = s;
  __syncthreads();
  if (threadIdx.x == 0) {
    float t = 0.f;
    for (int i = 0; i < 8; ++i) t += red[i];
    red[0] = t;
  }
  __syncthreads();
  float inv = 1.0f / red[0];
  o[threadIdx.x]       = (_Float16)(e0 * inv);
  o[threadIdx.x + 256] = (_Float16)(e1 * inv);
  o[threadIdx.x + 512] = (_Float16)(e2 * inv);
}

// ---------------------------------------------------------------------------
// Point frames: pts[n][hp][:] = rot[n] * praw[n][:,hp] + trans[n]
// ---------------------------------------------------------------------------
__global__ void points_kernel(const float* __restrict__ praw, const float* __restrict__ rot,
                              const float* __restrict__ trans, float* __restrict__ pts, int HP) {
  int idx = blockIdx.x * blockDim.x + threadIdx.x;
  int n = idx / HP, hp = idx % HP;
  if (n >= NRES) return;
  const float* R = rot + n * 9;
  const float* t = trans + n * 3;
  const float* pr = praw + (long)n * 3 * HP;
  float x = pr[hp], y = pr[HP + hp], zz = pr[2 * HP + hp];
  float* o = pts + ((long)n * HP + hp) * 3;
  o[0] = R[0] * x + R[1] * y + R[2] * zz + t[0];
  o[1] = R[3] * x + R[4] * y + R[5] * zz + t[1];
  o[2] = R[6] * x + R[7] * y + R[8] * zz + t[2];
}

// ---------------------------------------------------------------------------
// Epilogue: inverse-frame points, norms, concat -> f16 cat buffer
// ---------------------------------------------------------------------------
__global__ void cat_kernel(const float* __restrict__ o, const float* __restrict__ opt,
                           const float* __restrict__ opair, const float* __restrict__ rot,
                           const float* __restrict__ trans, _Float16* __restrict__ cat) {
  int n = blockIdx.x;
  const float* R = rot + n * 9;
  const float* t = trans + n * 3;
  _Float16* c = cat + (long)n * CATD;
  for (int i = threadIdx.x; i < NH * CHD; i += blockDim.x)
    c[i] = (_Float16)o[(long)n * NH * CHD + i];
  for (int j = threadIdx.x; j < NH * PVD; j += blockDim.x) {
    const float* p = opt + ((long)n * NH * PVD + j) * 3;
    float bx = p[0] - t[0], by = p[1] - t[1], bz = p[2] - t[2];
    float ax = R[0] * bx + R[3] * by + R[6] * bz;   // rot^T
    float ay = R[1] * bx + R[4] * by + R[7] * bz;
    float az = R[2] * bx + R[5] * by + R[8] * bz;
    c[192 + j] = (_Float16)ax;
    c[288 + j] = (_Float16)ay;
    c[384 + j] = (_Float16)az;
    c[480 + j] = (_Float16)sqrtf(ax * ax + ay * ay + az * az + 1e-8f);
  }
  for (int i = threadIdx.x; i < NH * CZD; i += blockDim.x)
    c[576 + i] = (_Float16)opair[(long)n * NH * CZD + i];
}

// ---------------------------------------------------------------------------
// Pack kernels
// ---------------------------------------------------------------------------
__global__ void pack_f16(const float* __restrict__ in, _Float16* __restrict__ out, long n) {
  long i = (long)blockIdx.x * blockDim.x + threadIdx.x;
  if (i < n) out[i] = (_Float16)in[i];
}

// W (KxN f32) -> Bt (Np x K f16), zero pad rows n >= N
__global__ void pack_wT(const float* __restrict__ W, _Float16* __restrict__ Bt,
                        int K, int N, int Np) {
  long i = (long)blockIdx.x * blockDim.x + threadIdx.x;
  if (i >= (long)Np * K) return;
  int nidx = (int)(i / K), k = (int)(i % K);
  Bt[i] = (nidx < N) ? (_Float16)W[(long)k * N + nidx] : (_Float16)0.f;
}

// qh/kh: (h, n, 32) f16 with K=16 zero-padded to 32
__global__ void pack_qk(const float* __restrict__ qbuf, const float* __restrict__ kvbuf,
                        _Float16* __restrict__ qh, _Float16* __restrict__ kh) {
  long i = (long)blockIdx.x * blockDim.x + threadIdx.x;
  if (i >= (long)NH * NRES * 32) return;
  int cc = (int)(i & 31);
  int r = (int)((i >> 5) % NRES);
  int h = (int)(i / (NRES * 32));
  qh[i] = (cc < CHD) ? (_Float16)qbuf[(long)r * (NH * CHD) + h * CHD + cc] : (_Float16)0.f;
  kh[i] = (cc < CHD) ? (_Float16)kvbuf[(long)r * (2 * NH * CHD) + h * 2 * CHD + cc] : (_Float16)0.f;
}

// vt: (h, c=16, k=768) f16 from kv[..., CH:]
__global__ void pack_vt(const float* __restrict__ kv, _Float16* __restrict__ vt) {
  long i = (long)blockIdx.x * blockDim.x + threadIdx.x;
  if (i >= (long)NH * CHD * NRES) return;
  int k = (int)(i % NRES);
  int c = (int)((i / NRES) % CHD);
  int h = (int)(i / (NRES * CHD));
  vt[i] = (_Float16)kv[(long)k * (2 * NH * CHD) + h * 2 * CHD + CHD + c];
}

// vptsT: (h, c=32, k=768) f16 from v_pts components (24 real cols, pad to 32)
__global__ void pack_vptsT(const float* __restrict__ kvpts, _Float16* __restrict__ vp) {
  long i = (long)blockIdx.x * blockDim.x + threadIdx.x;
  if (i >= (long)NH * 32 * NRES) return;
  int k = (int)(i % NRES);
  int c = (int)((i / NRES) % 32);
  int h = (int)(i / (NRES * 32));
  vp[i] = (c < PVD * 3)
            ? (_Float16)kvpts[(long)k * (NH * (PQKD + PVD) * 3) + h * (PQKD + PVD) * 3 + PQKD * 3 + c]
            : (_Float16)0.f;
}

// ---------------------------------------------------------------------------
// Host side
// ---------------------------------------------------------------------------
static inline void launch_gemm(hipStream_t st, const _Float16* A, const _Float16* Bt,
                               const float* bias, float* C,
                               int M, int N, int Kp, int lda, int ldbt, int ldc,
                               long sA, long sB, long sC, int batch) {
  int tiles = ((M + 15) / 16) * ((N + 15) / 16);
  dim3 grid((tiles + 7) / 8, batch);
  gemm_f16_wmma<<<grid, dim3(256), 0, st>>>(A, Bt, bias, C, M, N, Kp,
                                            lda, ldbt, ldc, sA, sB, sC);
}

extern "C" void kernel_launch(void* const* d_in, const int* in_sizes, int n_in,
                              void* d_out, int out_size, void* d_ws, size_t ws_size,
                              hipStream_t stream) {
  const float* s      = (const float*)d_in[0];
  const float* z      = (const float*)d_in[1];
  const float* rot    = (const float*)d_in[2];
  const float* trans  = (const float*)d_in[3];
  const float* mask   = (const float*)d_in[4];
  const float* W_q    = (const float*)d_in[5];
  const float* b_q    = (const float*)d_in[6];
  const float* W_kv   = (const float*)d_in[7];
  const float* b_kv   = (const float*)d_in[8];
  const float* W_qp   = (const float*)d_in[9];
  const float* b_qp   = (const float*)d_in[10];
  const float* W_kvp  = (const float*)d_in[11];
  const float* b_kvp  = (const float*)d_in[12];
  const float* W_b    = (const float*)d_in[13];
  const float* b_b    = (const float*)d_in[14];
  const float* head_w = (const float*)d_in[15];
  const float* W_out  = (const float*)d_in[16];
  const float* b_out  = (const float*)d_in[17];
  float* out = (float*)d_out;

  // Bump allocator over workspace (~67 MB total), 256B aligned
  char* wp = (char*)d_ws;
  auto alloc = [&](size_t bytes) { char* r = wp; wp += (bytes + 255) & ~(size_t)255; return r; };

  _Float16* s16    = (_Float16*)alloc((size_t)NRES * CSD * 2);
  _Float16* WqT    = (_Float16*)alloc((size_t)192 * CSD * 2);
  _Float16* WkvT   = (_Float16*)alloc((size_t)384 * CSD * 2);
  _Float16* WqpT   = (_Float16*)alloc((size_t)144 * CSD * 2);
  _Float16* WkvpT  = (_Float16*)alloc((size_t)432 * CSD * 2);
  _Float16* WbT    = (_Float16*)alloc((size_t)16 * CZD * 2);
  _Float16* WoutT  = (_Float16*)alloc((size_t)CSD * CATD * 2);
  float*    qbuf   = (float*)alloc((size_t)NRES * 192 * 4);
  float*    kvbuf  = (float*)alloc((size_t)NRES * 384 * 4);
  float*    qpraw  = (float*)alloc((size_t)NRES * 144 * 4);
  float*    kvpraw = (float*)alloc((size_t)NRES * 432 * 4);
  float*    qpts   = (float*)alloc((size_t)NRES * 144 * 4);
  float*    kvpts  = (float*)alloc((size_t)NRES * 432 * 4);
  _Float16* qh     = (_Float16*)alloc((size_t)NH * NRES * 32 * 2);
  _Float16* kh     = (_Float16*)alloc((size_t)NH * NRES * 32 * 2);
  _Float16* vt     = (_Float16*)alloc((size_t)NH * CHD * NRES * 2);
  _Float16* vptsT  = (_Float16*)alloc((size_t)NH * 32 * NRES * 2);
  float*    att    = (float*)alloc((size_t)NH * ATT_S * 4);
  _Float16* att16  = (_Float16*)alloc((size_t)16 * ATT_S * 2);   // 16 planes (M pad)
  float*    obuf   = (float*)alloc((size_t)NRES * 192 * 4);
  float*    optbuf = (float*)alloc((size_t)NRES * 288 * 4);
  float*    opair  = (float*)alloc((size_t)NRES * NH * CZD * 4);
  _Float16* cat16  = (_Float16*)alloc((size_t)NRES * CATD * 2);

  auto blks = [](long n) { return (unsigned)((n + 255) / 256); };

  // 0) Pack inputs/weights to f16 (Bt = N_pad x K, zero padded)
  pack_f16<<<blks((long)NRES * CSD), 256, 0, stream>>>(s, s16, (long)NRES * CSD);
  pack_wT<<<blks((long)192 * CSD), 256, 0, stream>>>(W_q, WqT, CSD, 192, 192);
  pack_wT<<<blks((long)384 * CSD), 256, 0, stream>>>(W_kv, WkvT, CSD, 384, 384);
  pack_wT<<<blks((long)144 * CSD), 256, 0, stream>>>(W_qp, WqpT, CSD, 144, 144);
  pack_wT<<<blks((long)432 * CSD), 256, 0, stream>>>(W_kvp, WkvpT, CSD, 432, 432);
  pack_wT<<<blks((long)16 * CZD), 256, 0, stream>>>(W_b, WbT, CZD, NH, 16);
  pack_wT<<<blks((long)CSD * CATD), 256, 0, stream>>>(W_out, WoutT, CATD, CSD, CSD);

  // 1) Projections from s (M=768, K=384)
  launch_gemm(stream, s16, WqT,   b_q,   qbuf,   NRES, 192, CSD, CSD, CSD, 192, 0, 0, 0, 1);
  launch_gemm(stream, s16, WkvT,  b_kv,  kvbuf,  NRES, 384, CSD, CSD, CSD, 384, 0, 0, 0, 1);
  launch_gemm(stream, s16, WqpT,  b_qp,  qpraw,  NRES, 144, CSD, CSD, CSD, 144, 0, 0, 0, 1);
  launch_gemm(stream, s16, WkvpT, b_kvp, kvpraw, NRES, 432, CSD, CSD, CSD, 432, 0, 0, 0, 1);

  // 2) Rotate/translate points into global frame
  points_kernel<<<blks((long)NRES * 48), 256, 0, stream>>>(qpraw, rot, trans, qpts, 48);
  points_kernel<<<blks((long)NRES * 144), 256, 0, stream>>>(kvpraw, rot, trans, kvpts, 144);

  // 3) Pack per-head operands
  pack_qk<<<blks((long)NH * NRES * 32), 256, 0, stream>>>(qbuf, kvbuf, qh, kh);
  pack_vt<<<blks((long)NH * CHD * NRES), 256, 0, stream>>>(kvbuf, vt);
  pack_vptsT<<<blks((long)NH * 32 * NRES), 256, 0, stream>>>(kvpts, vptsT);

  // 4) Pair bias (streams z once, non-temporal) -> att f32 in (h,q,k) layout
  bias_pair_kernel<<<(NRES * NRES / 16 + 7) / 8, 256, 0, stream>>>(z, WbT, b_b, att);

  // 5) qk^T + point attention + mask
  {
    const int KT = NRES / 16;
    logits_kernel<<<dim3((KT * KT) / 8, NH), 256, 0, stream>>>(qh, kh, qpts, kvpts,
                                                               mask, head_w, att);
  }

  // 6) Softmax over k -> f16 attention planes
  softmax_kernel<<<NH * NRES, 256, 0, stream>>>(att, att16);

  // 7) o = a.v  (batch over heads)
  launch_gemm(stream, att16, vt, nullptr, obuf,
              NRES, CHD, NRES, NRES, NRES, NH * CHD,
              ATT_S, (long)CHD * NRES, CHD, NH);
  // 8) o_pt = a.v_pts (batch over heads; 24 real cols)
  launch_gemm(stream, att16, vptsT, nullptr, optbuf,
              NRES, PVD * 3, NRES, NRES, NRES, NH * PVD * 3,
              ATT_S, (long)32 * NRES, PVD * 3, NH);
  // 9) o_pair = a.z (batch over q; streams z once more, non-temporal)
  opair_kernel<<<NRES, 256, 0, stream>>>(att16, z, opair);

  // 10) Inverse-frame points, norms, concat (f16)
  cat_kernel<<<NRES, 256, 0, stream>>>(obuf, optbuf, opair, rot, trans, cat16);

  // 11) out = cat @ W_out + b_out
  launch_gemm(stream, cat16, WoutT, b_out, out,
              NRES, CSD, CATD, CATD, CATD, CSD, 0, 0, 0, 1);
}
